// TransformerLayer_50525995270826
// MI455X (gfx1250) — matrix operand order repaired and checked by reading, not compiled, not run on hardware
//
#include <hip/hip_runtime.h>
#include <math.h>

// ---------------- problem constants ----------------
#define B_   2
#define T_   2048
#define C_   768
#define H_   12
#define FF_  3072
#define D_   64
#define BH_  (B_ * H_)
#define M_   (B_ * T_)

// ---------------- vector types ----------------
typedef __attribute__((ext_vector_type(16))) _Float16 v16h;
typedef __attribute__((ext_vector_type(8)))  _Float16 v8h;
typedef __attribute__((ext_vector_type(8)))  float    v8f;

// ---------------- WMMA helpers ----------------
__device__ __forceinline__ v8f wmma16(v16h a, v16h b, v8f c) {
  // D = A(16x32 f16) * B(32x16 f16) + C(16x16 f32)
  return __builtin_amdgcn_wmma_f32_16x16x32_f16(
      /*neg_a=*/false, a, /*neg_b=*/false, b,
      /*c_mod=*/(short)0, c, /*reuse_a=*/false, /*reuse_b=*/false);
}

// A-fragment: 16x32 tile at (m0,k0) of row-major [.,ld] f16 matrix.
// lane r=lane&15 holds row m0+r; elems i (0..15) -> k0+g*16+i (contig.)
__device__ __forceinline__ v16h load_a_frag(const _Float16* __restrict__ A,
                                            int ld, int m0, int k0) {
  int lane = threadIdx.x & 31;
  int r = lane & 15, g = lane >> 4;
  const _Float16* p = A + (size_t)(m0 + r) * ld + k0 + g * 16;
  v8h lo = *(const v8h*)p;
  v8h hi = *(const v8h*)(p + 8);
  v16h a;
#pragma unroll
  for (int i = 0; i < 8; ++i) { a[i] = lo[i]; a[i + 8] = hi[i]; }
  return a;
}

// B-fragment for D = A * W^T where W is row-major [N,K] f16.
// B[k][n] = W[n0+n][k0+k]; lane col n=lane&15, elems i -> k = g*16+i (contig.)
__device__ __forceinline__ v16h load_bT_frag(const _Float16* __restrict__ W,
                                             int ld, int n0, int k0) {
  int lane = threadIdx.x & 31;
  int n = lane & 15, g = lane >> 4;
  return *(const v16h*)(W + (size_t)(n0 + n) * ld + k0 + g * 16);
}

// ---------------- small kernels ----------------
__global__ __launch_bounds__(256)
void k_f32_to_f16(const float* __restrict__ in, _Float16* __restrict__ out, long n) {
  long i = (long)blockIdx.x * 256 + threadIdx.x;
  long stride = (long)gridDim.x * 256;
  for (; i < n; i += stride) out[i] = (_Float16)in[i];
}

__global__ __launch_bounds__(256)
void k_layernorm(const float* __restrict__ x, const float* __restrict__ w,
                 const float* __restrict__ b, _Float16* __restrict__ out) {
  __shared__ float red1[256];
  __shared__ float red2[256];
  int row = blockIdx.x;
  const float* xr = x + (size_t)row * C_;
  float s = 0.f, ss = 0.f;
  for (int c = threadIdx.x; c < C_; c += 256) { float v = xr[c]; s += v; ss += v * v; }
  red1[threadIdx.x] = s; red2[threadIdx.x] = ss;
  __syncthreads();
  for (int st = 128; st > 0; st >>= 1) {
    if ((int)threadIdx.x < st) {
      red1[threadIdx.x] += red1[threadIdx.x + st];
      red2[threadIdx.x] += red2[threadIdx.x + st];
    }
    __syncthreads();
  }
  float mean = red1[0] * (1.0f / C_);
  float var  = red2[0] * (1.0f / C_) - mean * mean;
  float rstd = rsqrtf(var + 1e-5f);
  for (int c = threadIdx.x; c < C_; c += 256)
    out[(size_t)row * C_ + c] = (_Float16)((xr[c] - mean) * rstd * w[c] + b[c]);
}

// g_in[b*D+d] = mean over (h,t) of q[b,h,t,d]
__global__ __launch_bounds__(256)
void k_qmean(const _Float16* __restrict__ q, float* __restrict__ gin) {
  __shared__ float red[256];
  int bd = blockIdx.x;          // B_*D_ blocks
  int b = bd / D_, d = bd % D_;
  float s = 0.f;
  for (int ht = threadIdx.x; ht < H_ * T_; ht += 256) {
    int hh = ht / T_, t = ht % T_;
    s += (float)q[((size_t)(b * H_ + hh) * T_ + t) * D_ + d];
  }
  red[threadIdx.x] = s;
  __syncthreads();
  for (int st = 128; st > 0; st >>= 1) {
    if ((int)threadIdx.x < st) red[threadIdx.x] += red[threadIdx.x + st];
    __syncthreads();
  }
  if (threadIdx.x == 0) gin[bd] = red[0] * (1.0f / (H_ * T_));
}

// gates[b*H+h] = sigmoid(gin.dot(i_w[h])) * rel_scale[h] * sigmoid(gin.dot(ur_w[h]))
__global__ void k_gates(const float* __restrict__ gin, const float* __restrict__ ur_w,
                        const float* __restrict__ i_w, const float* __restrict__ rs,
                        float* __restrict__ gates) {
  int t = threadIdx.x;
  if (t < BH_) {
    int b = t / H_, h = t % H_;
    float a = 0.f, c = 0.f;
    for (int d = 0; d < D_; ++d) {
      float g = gin[b * D_ + d];
      a += g * ur_w[h * D_ + d];
      c += g * i_w[h * D_ + d];
    }
    float gr = 1.f / (1.f + __expf(-a));
    float gu = 1.f / (1.f + __expf(-c));
    gates[t] = gu * rs[h] * gr;   // "+1" applied in attention kernel
  }
}

// peTab[(rel + T-1)*H + h] = rel_embed[bucket(rel)*H + h]
__global__ __launch_bounds__(256)
void k_petab(const float* __restrict__ rel_embed, float* __restrict__ peTab) {
  int idx = blockIdx.x * 256 + threadIdx.x;
  if (idx >= 2 * T_ - 1) return;
  int rel = idx - (T_ - 1);
  int sign = (rel >= 0) ? 1 : 0;
  int ab = rel < 0 ? -rel : rel;
  int bucket;
  if (ab < 80) {
    bucket = ab;
  } else {
    float lr = __logf(fmaxf((float)ab, 1.0f) * (1.0f / 80.f)) * (1.0f / __logf(10.0f));
    int lp = 80 + (int)(lr * 80.0f);   // trunc toward zero (lr >= 0 here)
    bucket = lp < 159 ? lp : 159;
  }
  bucket += sign * 160;
  bucket = bucket < 0 ? 0 : (bucket > 319 ? 319 : bucket);
  for (int h = 0; h < H_; ++h)
    peTab[(size_t)idx * H_ + h] = rel_embed[bucket * H_ + h];
}

// ---------------- WMMA GEMM: out = A[M,K] @ W[N,K]^T + bias ----------------
// Block = 4 waves (128 thr) computing a 128x128 tile; each wave owns 64x64 =
// 4x4 WMMA tiles -> 16 v_wmma per K-step against 8 fragment loads (32 FLOP/B).
enum { MODE_F32 = 0, MODE_GELU = 1, MODE_SCATTER = 2 };

__global__ __launch_bounds__(128)
void k_gemm(const _Float16* __restrict__ A, const _Float16* __restrict__ W,
            const float* __restrict__ bias, const float* __restrict__ resid,
            float* __restrict__ outF, _Float16* __restrict__ outH,
            int M, int N, int K, int mode) {
  int w  = threadIdx.x >> 5;
  int wr = w & 1, wc = w >> 1;                 // 2x2 wave grid
  int m_wave = blockIdx.y * 128 + wr * 64;
  int n_wave = blockIdx.x * 128 + wc * 64;

  v8f zero = {};
  v8f acc[4][4];
#pragma unroll
  for (int i = 0; i < 4; ++i)
#pragma unroll
    for (int j = 0; j < 4; ++j) acc[i][j] = zero;

  for (int k0 = 0; k0 < K; k0 += 32) {
    if (k0 + 32 < K) {
      __builtin_prefetch(A + (size_t)m_wave * K + k0 + 32, 0, 3);
      __builtin_prefetch(W + (size_t)n_wave * K + k0 + 32, 0, 3);
    }
    v16h af[4], bf[4];
#pragma unroll
    for (int ti = 0; ti < 4; ++ti) af[ti] = load_a_frag(A, K, m_wave + ti * 16, k0);
#pragma unroll
    for (int tj = 0; tj < 4; ++tj) bf[tj] = load_bT_frag(W, K, n_wave + tj * 16, k0);
#pragma unroll
    for (int ti = 0; ti < 4; ++ti)
#pragma unroll
      for (int tj = 0; tj < 4; ++tj)
        acc[ti][tj] = wmma16(af[ti], bf[tj], acc[ti][tj]);
  }

  int lane = threadIdx.x & 31;
  int n = lane & 15, g = lane >> 4;
#pragma unroll
  for (int ti = 0; ti < 4; ++ti)
#pragma unroll
    for (int tj = 0; tj < 4; ++tj)
#pragma unroll
      for (int v = 0; v < 8; ++v) {
        int row = m_wave + ti * 16 + g * 8 + v;
        int col = n_wave + tj * 16 + n;
        float val = acc[ti][tj][v];
        if (bias) val += bias[col];
        if (mode == MODE_F32) {
          float r = resid ? resid[(size_t)row * N + col] : 0.f;
          outF[(size_t)row * N + col] = val + r;
        } else if (mode == MODE_GELU) {
          outH[(size_t)row * N + col] =
              (_Float16)(0.5f * val * (1.0f + erff(val * 0.70710678f)));
        } else {  // MODE_SCATTER: [M,C] element -> [B,H,T,D]
          int bb = row >> 11, t = row & (T_ - 1);
          int hh = col >> 6,  d = col & (D_ - 1);
          outH[((size_t)(bb * H_ + hh) * T_ + t) * D_ + d] = (_Float16)val;
        }
      }
}

// ---------------- flash attention with gated rel-pos bias ----------------
// grid: (T/64, B*H); block: 128 (4 waves, 16 q-rows each)
__global__ __launch_bounds__(128)
void k_attn(const _Float16* __restrict__ Q, const _Float16* __restrict__ K,
            const _Float16* __restrict__ V, const float* __restrict__ peTab,
            const float* __restrict__ gates, _Float16* __restrict__ O) {
  __shared__ _Float16 lK [64 * 64];      // [kv][d]  row-major
  __shared__ _Float16 lVt[64 * 64];      // [d][kv]  transposed
  __shared__ _Float16 lP [4][16 * 64];   // per-wave P tile [m][kv]

  int bh = blockIdx.y;
  int b = bh / H_, h = bh % H_;
  int tid = threadIdx.x;
  int w = tid >> 5, lane = tid & 31;
  int ln = lane & 15, lg = lane >> 4;
  int qbase = blockIdx.x * 64 + w * 16;
  float gate = 1.0f + gates[bh];

  const _Float16* Qbh = Q + (size_t)bh * T_ * D_;
  const _Float16* Kbh = K + (size_t)bh * T_ * D_;
  const _Float16* Vbh = V + (size_t)bh * T_ * D_;
  _Float16* myP = lP[w];

  // Q fragments (kept in registers for the whole KV sweep)
  v16h qf[2];
  qf[0] = load_a_frag(Qbh, D_, qbase, 0);
  qf[1] = load_a_frag(Qbh, D_, qbase, 32);

  v8f zero = {};
  v8f o[4];
#pragma unroll
  for (int dt = 0; dt < 4; ++dt) o[dt] = zero;
  float mrow[8], lrow[8];
#pragma unroll
  for (int v = 0; v < 8; ++v) { mrow[v] = -1e30f; lrow[v] = 0.f; }

  for (int j0 = 0; j0 < T_; j0 += 64) {
    __syncthreads();  // previous tile fully consumed before overwrite
    // cooperative stage: K row-major, V transposed
    for (int idx = tid * 8; idx < 64 * 64; idx += 128 * 8) {
      int r = idx >> 6, c = idx & 63;
      *(v8h*)&lK[idx] = *(const v8h*)(Kbh + (size_t)(j0 + r) * D_ + c);
      v8h vv = *(const v8h*)(Vbh + (size_t)(j0 + r) * D_ + c);
#pragma unroll
      for (int i2 = 0; i2 < 8; ++i2) lVt[(c + i2) * 64 + r] = vv[i2];
    }
    __syncthreads();

    // S = Q @ K^T  (4 tiles of 16x16 over 64 kv columns)
    v8f s[4];
#pragma unroll
    for (int nt = 0; nt < 4; ++nt) {
      s[nt] = zero;
#pragma unroll
      for (int ks = 0; ks < 2; ++ks) {
        v16h bf = *(const v16h*)&lK[(nt * 16 + ln) * 64 + ks * 32 + lg * 16];
        s[nt] = wmma16(qf[ks], bf, s[nt]);
      }
    }

    // online softmax (per row m = lg*8+v; 16 columns live in one 16-lane group)
#pragma unroll
    for (int v = 0; v < 8; ++v) {
      int i = qbase + lg * 8 + v;
      float sv[4];
      float mx = -1e30f;
#pragma unroll
      for (int nt = 0; nt < 4; ++nt) {
        int j = j0 + nt * 16 + ln;
        float bias = peTab[(size_t)(j - i + (T_ - 1)) * H_ + h] * gate;
        float val = s[nt][v] * 0.125f + bias;     // scale = D^-0.5
        sv[nt] = val;
        mx = fmaxf(mx, val);
      }
#pragma unroll
      for (int off = 1; off < 16; off <<= 1) mx = fmaxf(mx, __shfl_xor(mx, off, 32));
      float mnew = fmaxf(mrow[v], mx);
      float corr = __expf(mrow[v] - mnew);
      mrow[v] = mnew;
      float psum = 0.f;
#pragma unroll
      for (int nt = 0; nt < 4; ++nt) {
        float p = __expf(sv[nt] - mnew);
        psum += p;
        myP[(lg * 8 + v) * 64 + nt * 16 + ln] = (_Float16)p;
      }
#pragma unroll
      for (int off = 1; off < 16; off <<= 1) psum += __shfl_xor(psum, off, 32);
      lrow[v] = lrow[v] * corr + psum;
#pragma unroll
      for (int dt = 0; dt < 4; ++dt) o[dt][v] *= corr;
    }

    // O += P @ V   (A from per-wave LDS tile, B from transposed V)
#pragma unroll
    for (int dt = 0; dt < 4; ++dt) {
#pragma unroll
      for (int ks = 0; ks < 2; ++ks) {
        const _Float16* pp = &myP[ln * 64 + ks * 32 + lg * 16];
        v8h lo = *(const v8h*)pp;
        v8h hi = *(const v8h*)(pp + 8);
        v16h af;
#pragma unroll
        for (int e = 0; e < 8; ++e) { af[e] = lo[e]; af[e + 8] = hi[e]; }
        v16h bf = *(const v16h*)&lVt[(dt * 16 + ln) * 64 + ks * 32 + lg * 16];
        o[dt] = wmma16(af, bf, o[dt]);
      }
    }
  }

  // normalize, scatter to [B,T,C] (c = h*D + d) as f16 for the out-projection
#pragma unroll
  for (int v = 0; v < 8; ++v) {
    int i = qbase + lg * 8 + v;
    float inv = 1.0f / lrow[v];
    size_t base = ((size_t)(b * T_ + i)) * C_ + h * D_;
#pragma unroll
    for (int dt = 0; dt < 4; ++dt)
      O[base + dt * 16 + ln] = (_Float16)(o[dt][v] * inv);
  }
}

// ---------------- launcher ----------------
extern "C" void kernel_launch(void* const* d_in, const int* in_sizes, int n_in,
                              void* d_out, int out_size, void* d_ws, size_t ws_size,
                              hipStream_t stream) {
  const float* x       = (const float*)d_in[0];
  const float* ln1_w   = (const float*)d_in[1];
  const float* ln1_b   = (const float*)d_in[2];
  const float* wq      = (const float*)d_in[3];
  const float* bq      = (const float*)d_in[4];
  const float* wk      = (const float*)d_in[5];
  const float* bk      = (const float*)d_in[6];
  const float* wv      = (const float*)d_in[7];
  const float* bv      = (const float*)d_in[8];
  const float* wo      = (const float*)d_in[9];
  const float* bo      = (const float*)d_in[10];
  const float* ln2_w   = (const float*)d_in[11];
  const float* ln2_b   = (const float*)d_in[12];
  const float* w1      = (const float*)d_in[13];
  const float* b1      = (const float*)d_in[14];
  const float* w2      = (const float*)d_in[15];
  const float* b2      = (const float*)d_in[16];
  const float* rel_emb = (const float*)d_in[17];
  const float* ur_w    = (const float*)d_in[18];
  const float* i_w     = (const float*)d_in[19];
  const float* rscale  = (const float*)d_in[20];
  float* out = (float*)d_out;

  // workspace carve (256B aligned)
  size_t off = 0;
  char* ws = (char*)d_ws;
  auto carve = [&](size_t bytes) -> char* {
    char* p = ws + off;
    off += (bytes + 255) & ~(size_t)255;
    return p;
  };
  _Float16* h1  = (_Float16*)carve((size_t)M_ * C_ * 2);
  _Float16* wqh = (_Float16*)carve((size_t)C_ * C_ * 2);
  _Float16* wkh = (_Float16*)carve((size_t)C_ * C_ * 2);
  _Float16* wvh = (_Float16*)carve((size_t)C_ * C_ * 2);
  _Float16* woh = (_Float16*)carve((size_t)C_ * C_ * 2);
  _Float16* w1h = (_Float16*)carve((size_t)FF_ * C_ * 2);
  _Float16* w2h = (_Float16*)carve((size_t)C_ * FF_ * 2);
  _Float16* qb  = (_Float16*)carve((size_t)BH_ * T_ * D_ * 2);
  _Float16* kb  = (_Float16*)carve((size_t)BH_ * T_ * D_ * 2);
  _Float16* vb  = (_Float16*)carve((size_t)BH_ * T_ * D_ * 2);
  _Float16* ao  = (_Float16*)carve((size_t)M_ * C_ * 2);
  float*    x2  = (float*)   carve((size_t)M_ * C_ * 4);
  _Float16* h2  = (_Float16*)carve((size_t)M_ * C_ * 2);
  _Float16* hff = (_Float16*)carve((size_t)M_ * FF_ * 2);
  float*    pe  = (float*)   carve((size_t)(2 * T_ - 1) * H_ * 4);
  float*    gin = (float*)   carve((size_t)B_ * D_ * 4);
  float*    gts = (float*)   carve((size_t)BH_ * 4);

  // 1) weights -> f16
  k_f32_to_f16<<<256, 256, 0, stream>>>(wq, wqh, (long)C_ * C_);
  k_f32_to_f16<<<256, 256, 0, stream>>>(wk, wkh, (long)C_ * C_);
  k_f32_to_f16<<<256, 256, 0, stream>>>(wv, wvh, (long)C_ * C_);
  k_f32_to_f16<<<256, 256, 0, stream>>>(wo, woh, (long)C_ * C_);
  k_f32_to_f16<<<256, 256, 0, stream>>>(w1, w1h, (long)FF_ * C_);
  k_f32_to_f16<<<256, 256, 0, stream>>>(w2, w2h, (long)C_ * FF_);

  // 2) LN1
  k_layernorm<<<M_, 256, 0, stream>>>(x, ln1_w, ln1_b, h1);

  // 3) QKV projections (scatter to [B,H,T,D]); 128x128 block tiles
  dim3 gCC(C_ / 128, M_ / 128);
  k_gemm<<<gCC, 128, 0, stream>>>(h1, wqh, bq, nullptr, nullptr, qb, M_, C_, C_, MODE_SCATTER);
  k_gemm<<<gCC, 128, 0, stream>>>(h1, wkh, bk, nullptr, nullptr, kb, M_, C_, C_, MODE_SCATTER);
  k_gemm<<<gCC, 128, 0, stream>>>(h1, wvh, bv, nullptr, nullptr, vb, M_, C_, C_, MODE_SCATTER);

  // 4) gates + rel-pos table
  k_qmean<<<B_ * D_, 256, 0, stream>>>(qb, gin);
  k_gates<<<1, 32, 0, stream>>>(gin, ur_w, i_w, rscale, gts);
  k_petab<<<(2 * T_ - 1 + 255) / 256, 256, 0, stream>>>(rel_emb, pe);

  // 5) attention
  k_attn<<<dim3(T_ / 64, BH_), 128, 0, stream>>>(qb, kb, vb, pe, gts, ao);

  // 6) out projection + residual -> x2 (f32)
  k_gemm<<<gCC, 128, 0, stream>>>(ao, woh, bo, x, x2, nullptr, M_, C_, C_, MODE_F32);

  // 7) LN2
  k_layernorm<<<M_, 256, 0, stream>>>(x2, ln2_w, ln2_b, h2);

  // 8) FF1 + GELU (f16), 9) FF2 + residual -> d_out
  k_gemm<<<dim3(FF_ / 128, M_ / 128), 128, 0, stream>>>(h2, w1h, b1, nullptr, nullptr, hff, M_, FF_, C_, MODE_GELU);
  k_gemm<<<gCC, 128, 0, stream>>>(hff, w2h, b2, x2, out, nullptr, M_, C_, FF_, MODE_F32);

  (void)in_sizes; (void)n_in; (void)out_size; (void)ws_size;
}
